// MultiHeadAttention_63342177682077
// MI455X (gfx1250) — compile-verified, matrix operand order checked by reading, not run on hardware
//
#include <hip/hip_runtime.h>
#include <cstdint>
#include <cstddef>

// ---------------------------------------------------------------------------
// MI455X (gfx1250) MHA block: bf16 WMMA GEMMs with async-to-LDS staging,
// TDM Q-tile load, flash-style softmax. B=4, S=1024, D=1024, H=16, DH=64.
// ---------------------------------------------------------------------------

#define BB 4
#define SS 1024
#define DDIM 1024
#define HH 16
#define DHH 64

typedef __attribute__((ext_vector_type(16))) __bf16 v16bf;
typedef __attribute__((ext_vector_type(8)))  float  v8f;
typedef __attribute__((ext_vector_type(4)))  float  v4f;
typedef __attribute__((ext_vector_type(4)))  unsigned u32x4;
typedef __attribute__((ext_vector_type(8)))  int      i32x8;
typedef __attribute__((ext_vector_type(4)))  int      i32x4;

union FragU { v16bf v; unsigned u[8]; };

// Generic pointers to LDS carry the LDS byte address in their low 32 bits
// (ISA 10.2).
__device__ __forceinline__ unsigned ldsoff(const void* p) {
  return (unsigned)(uintptr_t)p;
}

// Load a 16x32 bf16 WMMA A/B fragment from a K-contiguous buffer.
// ISA 7.12.2: lane L -> row (L&15); lo lanes K={0..7,16..23}, hi lanes
// K={8..15,24..31}; element pair 2g,2g+1 at k = 2(g&3)+16(g>>2)+8*(L>=16).
__device__ __forceinline__ v16bf load_frag(const __bf16* base, int row, int ldk) {
  const int lane = threadIdx.x & 31;
  const int r  = row + (lane & 15);
  const int kb = (lane >> 4) << 3;
  FragU f;
#pragma unroll
  for (int g = 0; g < 8; ++g) {
    const int k = ((g & 3) << 1) | ((g >> 2) << 4) | kb;
    f.u[g] = *reinterpret_cast<const unsigned*>(base + (size_t)r * ldk + k);
  }
  return f.v;
}

__device__ __forceinline__ v8f wmma_bf16(v16bf a, v16bf b, v8f c) {
  return __builtin_amdgcn_wmma_f32_16x16x32_bf16(false, a, false, b, (short)0, c,
                                                 false, false);
}

// Async-DMA one 128x64 bf16 tile (16 KB) global -> LDS. Tracked by ASYNCcnt.
__device__ __forceinline__ void stage_tile(unsigned ldsbase, uint64_t gbase,
                                           int rowbase, int ktElems, int t) {
#pragma unroll
  for (int r = 0; r < 4; ++r) {
    const int chunk = t + r * 256;                 // 0..1023 16B chunks
    const int row = chunk >> 3;                    // 8 chunks per 128B row
    const int col = (chunk & 7) << 3;              // bf16 elements
    const unsigned voff  = (unsigned)(((rowbase + row) * 1024 + ktElems + col) * 2);
    const unsigned laddr = ldsbase + (unsigned)chunk * 16;
    asm volatile("global_load_async_to_lds_b128 %0, %1, %2"
                 :
                 : "v"(laddr), "v"(voff), "s"(gbase)
                 : "memory");
  }
}

// TDM: load a 2-D bf16 tile (tile_dim0 x tile_dim1, row stride = 1024 elems)
// from global into LDS via the Tensor Data Mover (tracked by TENSORcnt).
// D# layout per ISA 8.3/8.4; groups 2/3 zero (<=2D tensor).
__device__ __forceinline__ void tdm_load_tile_2d(unsigned lds_addr, uint64_t gaddr,
                                                 int tile_d0, int tile_d1) {
  u32x4 g0;
  g0[0] = 1u;                                        // count=1, user descriptor
  g0[1] = lds_addr;                                  // LDS byte address
  g0[2] = (unsigned)gaddr;                           // global addr [31:0]
  g0[3] = (unsigned)((gaddr >> 32) & 0x1FFFFFFu) | (2u << 30);  // [56:32]|type=2
  i32x8 g1;
  g1[0] = 1 << 16;                                   // data_size=1 (2 bytes)
  g1[1] = (int)((DDIM & 0xFFFF) << 16);              // tensor_dim0[15:0]
  g1[2] = (int)(((BB * SS) & 0xFFFF) << 16);         // dim0[31:16]=0|dim1[15:0]
  g1[3] = (int)((tile_d0 & 0xFFFF) << 16);           // dim1[31:16]=0|tile_dim0
  g1[4] = tile_d1 & 0xFFFF;                          // tile_dim1 (tile_dim2=0)
  g1[5] = DDIM;                                      // tensor_dim0_stride lo32
  g1[6] = 0;                                         // stride0 hi | stride1 lo
  g1[7] = 0;
  i32x4 gz = {};
  i32x8 gz8 = {};
  __builtin_amdgcn_tensor_load_to_lds(g0, g1, gz, gz, gz8, 0);
}

// ---------------------------------------------------------------------------
// Elementwise f32 -> bf16 (activations).
// ---------------------------------------------------------------------------
__global__ __launch_bounds__(256)
void cvt_bf16_kernel(const float* __restrict__ in, __bf16* __restrict__ out) {
  const int i = (blockIdx.x * 256 + threadIdx.x) * 4;
  const v4f v = *reinterpret_cast<const v4f*>(in + i);
#pragma unroll
  for (int j = 0; j < 4; ++j) out[i + j] = (__bf16)v[j];
}

// ---------------------------------------------------------------------------
// Weight convert+transpose: W_f32[K][N] -> Wt_bf16[N][K]. 64x64 LDS tiles.
// ---------------------------------------------------------------------------
__global__ __launch_bounds__(256)
void cvtT_bf16_kernel(const float* __restrict__ in, __bf16* __restrict__ out) {
  __shared__ float tile[64][65];
  const int t = threadIdx.x;
  const int k0 = blockIdx.y * 64, n0 = blockIdx.x * 64;
  const int r = t >> 2, c0 = (t & 3) << 4;
#pragma unroll
  for (int j = 0; j < 16; ++j)
    tile[r][c0 + j] = in[(size_t)(k0 + r) * DDIM + n0 + c0 + j];
  __syncthreads();
#pragma unroll
  for (int j = 0; j < 16; ++j)
    out[(size_t)(n0 + r) * DDIM + k0 + c0 + j] = (__bf16)tile[c0 + j][r];
}

// ---------------------------------------------------------------------------
// Unified bf16 WMMA GEMM: C = A[4096x1024] @ Bt^T + bias, Bt stored [N][K].
// mode 0: bf16 row-major; mode 1: bf16 per-head transposed (V);
// mode 2: f32 + residual (output projection).
// Tile 128x128, BK=64, double-buffered async-to-LDS staging, 8 waves (4x2).
// ---------------------------------------------------------------------------
__global__ __launch_bounds__(256)
void gemm_bf16_kernel(const __bf16* __restrict__ A, const __bf16* __restrict__ Bt,
                      const float* __restrict__ bias,
                      __bf16* __restrict__ outb, float* __restrict__ outf,
                      const float* __restrict__ resid, int mode) {
  __shared__ __align__(16) __bf16 sA[2][128 * 64];
  __shared__ __align__(16) __bf16 sB[2][128 * 64];
  const int t = threadIdx.x, lane = t & 31, wave = t >> 5;
  const int wm = wave >> 1, wn = wave & 1;
  const int m0 = blockIdx.y * 128, n0 = blockIdx.x * 128;
  const uint64_t ga = (uint64_t)(uintptr_t)A;
  const uint64_t gb = (uint64_t)(uintptr_t)Bt;
  const unsigned la0 = ldsoff(sA[0]), la1 = ldsoff(sA[1]);
  const unsigned lb0 = ldsoff(sB[0]), lb1 = ldsoff(sB[1]);

  v8f acc[2][4] = {};

  // 2-deep async pipeline: tiles 0 and 1 in flight (16 copies/thread).
  stage_tile(la0, ga, m0, 0, t);
  stage_tile(lb0, gb, n0, 0, t);
  stage_tile(la1, ga, m0, 64, t);
  stage_tile(lb1, gb, n0, 64, t);

  for (int kt = 0; kt < DDIM; kt += 64) {
    const int cur = (kt >> 6) & 1;
    // ASYNCcnt retires in order: <=8 means the current tile's 8 copies done.
    if (kt == DDIM - 64) asm volatile("s_wait_asynccnt 0x0" ::: "memory");
    else                 asm volatile("s_wait_asynccnt 0x8" ::: "memory");
    __syncthreads();

    const __bf16* pa = sA[cur];
    const __bf16* pb = sB[cur];
#pragma unroll
    for (int ks = 0; ks < 64; ks += 32) {
      v16bf af[2], bfr[4];
#pragma unroll
      for (int i = 0; i < 2; ++i) af[i] = load_frag(pa + ks, wm * 32 + i * 16, 64);
#pragma unroll
      for (int j = 0; j < 4; ++j) bfr[j] = load_frag(pb + ks, wn * 64 + j * 16, 64);
#pragma unroll
      for (int i = 0; i < 2; ++i)
#pragma unroll
        for (int j = 0; j < 4; ++j) acc[i][j] = wmma_bf16(af[i], bfr[j], acc[i][j]);
    }
    __syncthreads();                   // all waves done reading buffer `cur`
    if (kt + 128 < DDIM) {             // refill it with tile kt+2
      stage_tile(cur ? la1 : la0, ga, m0, kt + 128, t);
      stage_tile(cur ? lb1 : lb0, gb, n0, kt + 128, t);
    }
  }

  // epilogue: C/D layout M = r + 8*(lane>=16), N = lane&15
#pragma unroll
  for (int i = 0; i < 2; ++i)
#pragma unroll
    for (int j = 0; j < 4; ++j) {
      const int nc = n0 + wn * 64 + j * 16 + (lane & 15);
      const float bv = bias[nc];
#pragma unroll
      for (int r = 0; r < 8; ++r) {
        const int m = m0 + wm * 32 + i * 16 + r + ((lane >> 4) << 3);
        const float c = acc[i][j][r] + bv;
        if (mode == 2) {
          outf[(size_t)m * DDIM + nc] = c + resid[(size_t)m * DDIM + nc];
        } else if (mode == 1) {
          const int bb = m >> 10, s = m & 1023;
          const int hh = nc >> 6, dh = nc & 63;
          outb[(((size_t)(bb * HH + hh) * DHH + dh) << 10) + s] = (__bf16)c;
        } else {
          outb[(size_t)m * DDIM + nc] = (__bf16)c;
        }
      }
    }
}

// ---------------------------------------------------------------------------
// Attention: one block per (b, h, 16-query tile). Q tile staged by the TDM;
// 8 waves each own 128 keys; two-pass flash softmax; probs -> d_out (f32)
// + LDS bf16 for context WMMA; ds_add_f32 cross-wave reduction.
// ---------------------------------------------------------------------------
__global__ __launch_bounds__(256)
void attention_kernel(const __bf16* __restrict__ Qbf, const __bf16* __restrict__ Kbf,
                      const __bf16* __restrict__ Vt, const unsigned char* __restrict__ mask,
                      float* __restrict__ attn, __bf16* __restrict__ ctx) {
  __shared__ __align__(16) __bf16 sQ[16 * 64];
  __shared__ __bf16 sP[16 * 1024];
  __shared__ float  sC[16 * 64];
  __shared__ float  swm[8 * 16], swl[8 * 16];
  __shared__ float  gM[16], gLinv[16];

  const int t = threadIdx.x, lane = t & 31, wave = t >> 5;
  const int bid = blockIdx.x;
  const int qt = bid & 63, h = (bid >> 6) & 15, b = bid >> 10;
  const int q0 = qt << 4;

#pragma unroll
  for (int i = 0; i < 4; ++i) sC[t + i * 256] = 0.f;

  // TDM: 16x64 bf16 Q tile, row stride D. Wave 0 issues, waits TENSORcnt,
  // then the workgroup barrier publishes it.
  if (wave == 0) {
    const uint64_t gq = (uint64_t)(uintptr_t)(Qbf + (size_t)(b * SS + q0) * DDIM +
                                              h * DHH);
    tdm_load_tile_2d(ldsoff(sQ), gq, DHH, 16);
    __builtin_amdgcn_s_wait_tensorcnt(0);
  }
  __syncthreads();

  const __bf16* Khead = Kbf + (size_t)b * SS * DDIM + h * DHH;
  const unsigned char* mrow = mask + (size_t)b * SS * SS;
  const float scale = 0.03125f;  // 1/sqrt(D)=1/32 (model-dim scaling quirk)

  float rmax[8], rsum[8];
#pragma unroll
  for (int r = 0; r < 8; ++r) { rmax[r] = -3.0e38f; rsum[r] = 0.f; }

  // ---- pass 1: online row max / sum ----
  for (int tile = 0; tile < 8; ++tile) {
    const int kb0 = wave * 128 + tile * 16;
    v8f acc = {};
#pragma unroll
    for (int kk = 0; kk < 64; kk += 32) {
      v16bf a = load_frag(sQ + kk, 0, 64);
      v16bf kfr = load_frag(Khead + kk, kb0, DDIM);
      acc = wmma_bf16(a, kfr, acc);
    }
    const int col = kb0 + (lane & 15);
    const int rloc = (lane >> 4) << 3;
#pragma unroll
    for (int r = 0; r < 8; ++r) {
      float s = acc[r] * scale;
      if (mrow[(size_t)(q0 + rloc + r) * SS + col]) s = -1e-9f;  // MASK_FILL
      float tm = s;
#pragma unroll
      for (int off = 1; off < 16; off <<= 1) tm = fmaxf(tm, __shfl_xor(tm, off, 32));
      const float nm = fmaxf(rmax[r], tm);
      float e = __expf(s - nm);
#pragma unroll
      for (int off = 1; off < 16; off <<= 1) e += __shfl_xor(e, off, 32);
      rsum[r] = rsum[r] * __expf(rmax[r] - nm) + e;
      rmax[r] = nm;
    }
  }
  if ((lane & 15) == 0) {
    const int half = lane >> 4;
#pragma unroll
    for (int r = 0; r < 8; ++r) {
      swm[wave * 16 + half * 8 + r] = rmax[r];
      swl[wave * 16 + half * 8 + r] = rsum[r];
    }
  }
  __syncthreads();
  if (t < 16) {
    float M = -3.0e38f;
    for (int w = 0; w < 8; ++w) M = fmaxf(M, swm[w * 16 + t]);
    float L = 0.f;
    for (int w = 0; w < 8; ++w) L += swl[w * 16 + t] * __expf(swm[w * 16 + t] - M);
    gM[t] = M;
    gLinv[t] = 1.f / L;
  }
  __syncthreads();

  // ---- pass 2: recompute scores, emit probabilities ----
  float* arow = attn + ((size_t)(b * HH + h) * SS + q0) * SS;
  for (int tile = 0; tile < 8; ++tile) {
    const int kb0 = wave * 128 + tile * 16;
    v8f acc = {};
#pragma unroll
    for (int kk = 0; kk < 64; kk += 32) {
      v16bf a = load_frag(sQ + kk, 0, 64);
      v16bf kfr = load_frag(Khead + kk, kb0, DDIM);
      acc = wmma_bf16(a, kfr, acc);
    }
    const int col = kb0 + (lane & 15);
    const int rloc = (lane >> 4) << 3;
#pragma unroll
    for (int r = 0; r < 8; ++r) {
      float s = acc[r] * scale;
      if (mrow[(size_t)(q0 + rloc + r) * SS + col]) s = -1e-9f;
      const float p = __expf(s - gM[rloc + r]) * gLinv[rloc + r];
      arow[(size_t)(rloc + r) * SS + col] = p;
      sP[(rloc + r) * 1024 + col] = (__bf16)p;
    }
  }
  __syncthreads();

  // ---- context: C[16x64] += P[16x1024] @ V[1024x64] ----
  {
    v8f cacc[4] = {};
    const __bf16* vhead = Vt + (size_t)(b * HH + h) * DHH * SS;  // [dh][token]
#pragma unroll
    for (int kc = 0; kc < 4; ++kc) {
      const int kbase = wave * 128 + kc * 32;
      v16bf a = load_frag(sP + kbase, 0, 1024);
#pragma unroll
      for (int j = 0; j < 4; ++j) {
        v16bf vfr = load_frag(vhead + kbase, j * 16, SS);
        cacc[j] = wmma_bf16(a, vfr, cacc[j]);
      }
    }
    const int rloc = (lane >> 4) << 3;
#pragma unroll
    for (int j = 0; j < 4; ++j) {
      const int cc = j * 16 + (lane & 15);
#pragma unroll
      for (int r = 0; r < 8; ++r) atomicAdd(&sC[(rloc + r) * 64 + cc], cacc[j][r]);
    }
  }
  __syncthreads();
  {
    __bf16* gc = ctx + (size_t)(b * SS + q0) * DDIM + h * DHH;
#pragma unroll
    for (int i = 0; i < 4; ++i) {
      const int idx = t + i * 256;
      gc[(size_t)(idx >> 6) * DDIM + (idx & 63)] = (__bf16)sC[idx];
    }
  }
}

// ---------------------------------------------------------------------------
// LayerNorm over rows of 1024.
// ---------------------------------------------------------------------------
__global__ __launch_bounds__(256)
void ln_kernel(const float* __restrict__ x, const float* __restrict__ g,
               const float* __restrict__ bta, float* __restrict__ out) {
  const int row = blockIdx.x, t = threadIdx.x;
  const float* xr = x + (size_t)row * DDIM;
  float v[4], s = 0.f, s2 = 0.f;
#pragma unroll
  for (int i = 0; i < 4; ++i) {
    v[i] = xr[t + i * 256];
    s += v[i];
    s2 += v[i] * v[i];
  }
#pragma unroll
  for (int off = 16; off; off >>= 1) {
    s  += __shfl_xor(s, off, 32);
    s2 += __shfl_xor(s2, off, 32);
  }
  __shared__ float ws1[8], ws2[8];
  __shared__ float mu_s, rstd_s;
  if ((t & 31) == 0) { ws1[t >> 5] = s; ws2[t >> 5] = s2; }
  __syncthreads();
  if (t == 0) {
    float a = 0.f, b2 = 0.f;
    for (int w = 0; w < 8; ++w) { a += ws1[w]; b2 += ws2[w]; }
    const float mu = a * (1.f / DDIM);
    const float var = b2 * (1.f / DDIM) - mu * mu;
    mu_s = mu;
    rstd_s = rsqrtf(var + 1e-5f);
  }
  __syncthreads();
#pragma unroll
  for (int i = 0; i < 4; ++i) {
    const int c = t + i * 256;
    out[(size_t)row * DDIM + c] = (v[i] - mu_s) * rstd_s * g[c] + bta[c];
  }
}

// ---------------------------------------------------------------------------
extern "C" void kernel_launch(void* const* d_in, const int* in_sizes, int n_in,
                              void* d_out, int out_size, void* d_ws, size_t ws_size,
                              hipStream_t stream) {
  (void)in_sizes; (void)n_in; (void)out_size; (void)ws_size;
  const float* query = (const float*)d_in[0];
  const float* key   = (const float*)d_in[1];
  const float* value = (const float*)d_in[2];
  const unsigned char* mask = (const unsigned char*)d_in[3];
  const float* Wq = (const float*)d_in[4];
  const float* bq = (const float*)d_in[5];
  const float* Wk = (const float*)d_in[6];
  const float* bk = (const float*)d_in[7];
  const float* Wv = (const float*)d_in[8];
  const float* bv = (const float*)d_in[9];
  const float* Wo = (const float*)d_in[10];
  const float* bo = (const float*)d_in[11];
  const float* ln_g = (const float*)d_in[12];
  const float* ln_b = (const float*)d_in[13];

  char* ws = (char*)d_ws;
  __bf16* qx  = (__bf16*)(ws);                        // 8 MB activations bf16
  __bf16* kx  = (__bf16*)(ws + ((size_t)8  << 20));   // 8 MB
  __bf16* vx  = (__bf16*)(ws + ((size_t)16 << 20));   // 8 MB
  __bf16* Wqt = (__bf16*)(ws + ((size_t)24 << 20));   // 2 MB [N][K]
  __bf16* Wkt = (__bf16*)(ws + ((size_t)26 << 20));   // 2 MB
  __bf16* Wvt = (__bf16*)(ws + ((size_t)28 << 20));   // 2 MB
  __bf16* Wot = (__bf16*)(ws + ((size_t)30 << 20));   // 2 MB
  __bf16* Qbf = (__bf16*)(ws + ((size_t)32 << 20));   // 8 MB
  __bf16* Kbf = (__bf16*)(ws + ((size_t)40 << 20));   // 8 MB
  __bf16* VtT = (__bf16*)(ws + ((size_t)48 << 20));   // 8 MB per-head [dh][s]
  __bf16* Ctx = (__bf16*)(ws + ((size_t)56 << 20));   // 8 MB
  float*  xbf = (float*)(ws + ((size_t)64 << 20));    // 16 MB

  float* outp = (float*)d_out;
  float* attn = outp + (size_t)BB * SS * DDIM;

  const int cvtBlocks = (BB * SS * DDIM) / 1024;      // 4 elems/thread
  cvt_bf16_kernel<<<cvtBlocks, 256, 0, stream>>>(query, qx);
  cvt_bf16_kernel<<<cvtBlocks, 256, 0, stream>>>(key,   kx);
  cvt_bf16_kernel<<<cvtBlocks, 256, 0, stream>>>(value, vx);
  const dim3 tg(16, 16);
  cvtT_bf16_kernel<<<tg, 256, 0, stream>>>(Wq, Wqt);
  cvtT_bf16_kernel<<<tg, 256, 0, stream>>>(Wk, Wkt);
  cvtT_bf16_kernel<<<tg, 256, 0, stream>>>(Wv, Wvt);
  cvtT_bf16_kernel<<<tg, 256, 0, stream>>>(Wo, Wot);

  const dim3 gg(8, 32), blk(256);
  gemm_bf16_kernel<<<gg, blk, 0, stream>>>(qx, Wqt, bq, Qbf, nullptr, nullptr, 0);
  gemm_bf16_kernel<<<gg, blk, 0, stream>>>(kx, Wkt, bk, Kbf, nullptr, nullptr, 0);
  gemm_bf16_kernel<<<gg, blk, 0, stream>>>(vx, Wvt, bv, VtT, nullptr, nullptr, 1);
  attention_kernel<<<BB * HH * (SS / 16), 256, 0, stream>>>(Qbf, Kbf, VtT, mask,
                                                            attn, Ctx);
  gemm_bf16_kernel<<<gg, blk, 0, stream>>>(Ctx, Wot, bo, nullptr, xbf, query, 2);
  ln_kernel<<<BB * SS, 256, 0, stream>>>(xbf, ln_g, ln_b, outp);
}
